// HDCBinaryAttention_59863254172674
// MI455X (gfx1250) — 16-bit and 8-bit matrix operand order repaired and checked by reading, compiled, not run on hardware
//
#include <hip/hip_runtime.h>


#define B_ 4
#define T_ 2048
#define D_ 1024

#define SSTR 68
#define PSTR 72

typedef __attribute__((ext_vector_type(16))) _Float16 v16h;
typedef __attribute__((ext_vector_type(8)))  _Float16 v8h;
typedef __attribute__((ext_vector_type(4)))  _Float16 v4h;
typedef __attribute__((ext_vector_type(8)))  float    v8f;
typedef __attribute__((ext_vector_type(4)))  float    v4f;
typedef __attribute__((ext_vector_type(16))) int      v16i;
typedef __attribute__((ext_vector_type(4)))  int      v4i;

__device__ __forceinline__ float sgnf(float v) {
    return (v > 0.0f) ? 1.0f : ((v < 0.0f) ? -1.0f : 0.0f);
}

__device__ __forceinline__ unsigned char fp8_sign(float v) {
    return (v > 0.0f) ? 0x38u : ((v < 0.0f) ? 0xB8u : 0x00u);
}

__global__ __launch_bounds__(256)
void hdc_prep_kernel(const float* __restrict__ x,
                     const float* __restrict__ bvq,
                     const float* __restrict__ bvk,
                     const float* __restrict__ bvv,
                     unsigned char* __restrict__ Qq,
                     unsigned char* __restrict__ Kq,
                     _Float16*      __restrict__ Vt)
{
    const size_t idx = (size_t)blockIdx.x * 256u + threadIdx.x;
    const int d = (int)(idx & (size_t)(D_ - 1));
    const int t = (int)((idx >> 10) & (size_t)(T_ - 1));
    const int b = (int)(idx >> 21);

    const float xv = x[idx];
    Qq[idx] = fp8_sign(xv * sgnf(bvq[d]));
    Kq[idx] = fp8_sign(xv * sgnf(bvk[d]));
    Vt[((size_t)b * D_ + d) * T_ + t] =
        (_Float16)(xv * sgnf(bvv[d]));
}

__global__ __launch_bounds__(256)
void hdc_attn_kernel(const unsigned char* __restrict__ Qq,
                     const unsigned char* __restrict__ Kq,
                     const _Float16*      __restrict__ Vt,
                     float*               __restrict__ out)
{
    __shared__ __attribute__((aligned(16))) float    S_lds[16 * SSTR];
    __shared__ __attribute__((aligned(16))) _Float16 P_lds[16 * PSTR];

    const int t0      = blockIdx.x << 4;
    const int b       = blockIdx.y;
    const int tid     = threadIdx.x;
    const int w       = tid >> 5;
    const int lane    = tid & 31;
    const int lane_lo = lane & 15;
    const int lane_hi = lane >> 4;
    const int dbase   = w << 7;

    v16i qa;
    {
        const size_t qrow = ((size_t)b * T_ + t0 + lane_lo) * D_;
#pragma unroll
        for (int v = 0; v < 16; ++v) {
            const int vv = v & 7;
            const int k0 = ((v >> 2) << 5) + (lane_hi << 4)
                         + ((v & 3) << 2);
            qa[v] = *(const int*)(Qq + qrow + dbase + k0);
        }
    }

    v8f acc[8];
#pragma unroll
    for (int n = 0; n < 8; ++n)
#pragma unroll
        for (int r = 0; r < 8; ++r) acc[n][r] = 0.0f;

    const int prow = tid >> 4;
    const int pcol = (tid & 15) << 2;

    const v8f zero8 = {0.f, 0.f, 0.f, 0.f, 0.f, 0.f, 0.f, 0.f};

    for (int s0 = 0; s0 < t0 + 16; s0 += 64) {
#pragma unroll
        for (int i = tid; i < 16 * SSTR; i += 256) S_lds[i] = 0.0f;
        __syncthreads();

        {
            v16i kb[4];
#pragma unroll
            for (int j = 0; j < 4; ++j) {
                const unsigned char* kp =
                    Kq + ((size_t)b * T_ + s0 + (j << 4) + lane_lo) * D_
                       + dbase + (lane_hi << 4);
                v16i f;
#pragma unroll
                for (int g = 0; g < 4; ++g) {
                    const v4i kg = *(const v4i*)(kp + (g << 5));
                    f[4 * g + 0] = kg[0]; f[4 * g + 1] = kg[1];
                    f[4 * g + 2] = kg[2]; f[4 * g + 3] = kg[3];
                }
                kb[j] = f;
            }
            __builtin_amdgcn_sched_barrier(0);
#pragma unroll
            for (int j = 0; j < 4; ++j) {
                const v8f s = __builtin_amdgcn_wmma_f32_16x16x128_fp8_fp8(
                                  qa, kb[j], (short)0, zero8, false, false);
#pragma unroll
                for (int r = 0; r < 8; ++r)
                    atomicAdd(&S_lds[(r + (lane_hi << 3)) * SSTR
                                     + (j << 4) + lane_lo], s[r]);
            }
        }
        __syncthreads();

        {
            const v4f sv4 = *(const v4f*)&S_lds[prow * SSTR + pcol];
            const int tg  = t0 + prow;
            v4h ph;
#pragma unroll
            for (int i = 0; i < 4; ++i) {
                const float e = __expf(-0.125f * sv4[i]);
                float p = __builtin_amdgcn_rcpf(1.0f + e);
                p = ((s0 + pcol + i) <= tg) ? p : 0.0f;
                ph[i] = (_Float16)p;
            }
            *(v4h*)&P_lds[prow * PSTR + pcol] = ph;
        }
        __syncthreads();

#pragma unroll
        for (int c = 0; c < 2; ++c) {
            const _Float16* pp =
                &P_lds[lane_lo * PSTR + (c << 5) + (lane_hi << 4)];
            const v8h plo = *(const v8h*)pp;
            const v8h phi = *(const v8h*)(pp + 8);
            v16h pa;
#pragma unroll
            for (int i = 0; i < 8; ++i) { pa[i] = plo[i]; pa[8 + i] = phi[i]; }

            v16h vb[8];
#pragma unroll
            for (int n = 0; n < 8; ++n)
                vb[n] = *(const v16h*)(
                    Vt + ((size_t)b * D_ + dbase + (n << 4) + lane_lo) * T_
                       + s0 + (c << 5) + (lane_hi << 4));
            __builtin_amdgcn_sched_barrier(0);
#pragma unroll
            for (int n = 0; n < 8; ++n)
                acc[n] = __builtin_amdgcn_wmma_f32_16x16x32_f16(
                             false, pa, false, vb[n], (short)0, acc[n],
                             false, false);
        }
    }

#pragma unroll
    for (int n = 0; n < 8; ++n) {
        float* op = out + ((size_t)b * T_ + t0 + (lane_hi << 3)) * D_
                        + dbase + (n << 4) + lane_lo;
#pragma unroll
        for (int r = 0; r < 8; ++r)
            op[(size_t)r * D_] = acc[n][r];
    }
}

extern "C" void kernel_launch(void* const* d_in, const int* in_sizes, int n_in,
                              void* d_out, int out_size, void* d_ws, size_t ws_size,
                              hipStream_t stream) {
    const float* x   = (const float*)d_in[0];
    const float* bvq = (const float*)d_in[1];
    const float* bvk = (const float*)d_in[2];
    const float* bvv = (const float*)d_in[3];
    float* out = (float*)d_out;

    const size_t n_elem = (size_t)B_ * T_ * D_;
    unsigned char* Qq = (unsigned char*)d_ws;
    unsigned char* Kq = Qq + n_elem;
    _Float16*      Vt = (_Float16*)(Kq + n_elem);

    hdc_prep_kernel<<<dim3((unsigned)(n_elem / 256)), dim3(256), 0, stream>>>(
        x, bvq, bvk, bvv, Qq, Kq, Vt);

    hdc_attn_kernel<<<dim3(T_ / 16, B_), dim3(256), 0, stream>>>(
        Qq, Kq, Vt, out);
}
